// VectorQuantizer_67894843015608
// MI455X (gfx1250) — matrix operand order repaired and checked by reading, not compiled, not run on hardware
//
#include <hip/hip_runtime.h>
#include <math.h>

// ---------------- CDNA5 WMMA types ----------------
typedef __attribute__((ext_vector_type(16))) __bf16  v16bf;
typedef __attribute__((ext_vector_type(8)))  float   v8f;
typedef __attribute__((ext_vector_type(4)))  unsigned int v4u;

union BFrag { v16bf v; unsigned short u[16]; v4u q[2]; };

// ---------------- problem constants ----------------
#define NVEC    32768      // 32*32*32 input vectors
#define KCB     1024       // codebook size
#define DIM     64         // embed dim
#define ROWS    32         // rows per block
#define DSTRIDE 1025       // padded LDS row stride (bank-conflict-free column scans)
#define THREADS 128        // 4 wave32s
#define NBLK    (NVEC / ROWS)
#define INV_TEMP 100.0f    // 1 / ENTROPY_TEMP

__device__ __forceinline__ unsigned short f32_to_bf16_rn(float f) {
  unsigned u = __float_as_uint(f);
  unsigned r = u + 0x7FFFu + ((u >> 16) & 1u);   // round-to-nearest-even
  return (unsigned short)(r >> 16);
}
__device__ __forceinline__ float bf16_bits_to_f32(unsigned short h) {
  return __uint_as_float(((unsigned)h) << 16);
}

// ============ kernel 1: codebook prep (bf16 hi/lo split + ||c||^2) ============
__global__ void vq_prep(const float* __restrict__ cb,
                        unsigned short* __restrict__ cbhi,
                        unsigned short* __restrict__ cblo,
                        float* __restrict__ c2) {
  int row = blockIdx.x * blockDim.x + threadIdx.x;
  if (row < KCB) {
    float s = 0.f;
    for (int d = 0; d < DIM; ++d) {
      float f = cb[(size_t)row * DIM + d];
      s = fmaf(f, f, s);
      unsigned short hi = f32_to_bf16_rn(f);
      float lo = f - bf16_bits_to_f32(hi);
      cbhi[(size_t)row * DIM + d] = hi;
      cblo[(size_t)row * DIM + d] = f32_to_bf16_rn(lo);
    }
    c2[row] = s;
  }
}

// ============ kernel 2: fused distances(WMMA) + argmin + softmax stats ============
__global__ void __launch_bounds__(THREADS)
vq_main(const float* __restrict__ xin, const float* __restrict__ cbf32,
        const unsigned short* __restrict__ cbhi, const unsigned short* __restrict__ cblo,
        const float* __restrict__ c2g,
        float* __restrict__ out, float* __restrict__ avgpart,
        float* __restrict__ entpart, float* __restrict__ msepart) {
  __shared__ float          s_dist[ROWS * DSTRIDE];   // distances, then exp() in place
  __shared__ unsigned short s_xhi[ROWS * DIM];
  __shared__ unsigned short s_xlo[ROWS * DIM];
  __shared__ float          s_c2[KCB];
  __shared__ float          s_x2[ROWS];
  __shared__ float          s_avgp[KCB];
  __shared__ float          s_ra[THREADS];
  __shared__ float          s_rb[THREADS];
  __shared__ int            s_ri[THREADS];
  __shared__ int            s_tok[ROWS];
  __shared__ float          s_amax[ROWS];
  __shared__ float          s_invS[ROWS];
  __shared__ float          s_entp[ROWS];

  const int t = threadIdx.x;
  const int blk = blockIdx.x;
  const int rowg0 = blk * ROWS;

  // ---- phase 1: stage x (bf16 hi/lo split) + row norms + c2 into LDS ----
  for (int i = t; i < KCB; i += THREADS) s_c2[i] = c2g[i];
  if (t < ROWS) {
    const float* xr = xin + (size_t)(rowg0 + t) * DIM;
    float s = 0.f;
    for (int d = 0; d < DIM; ++d) {
      float f = xr[d];
      s = fmaf(f, f, s);
      unsigned short hi = f32_to_bf16_rn(f);
      float lo = f - bf16_bits_to_f32(hi);
      s_xhi[t * DIM + d] = hi;
      s_xlo[t * DIM + d] = f32_to_bf16_rn(lo);
    }
    s_x2[t] = s;
  }
  __syncthreads();

  // ---- phase 2: distances via v_wmma_f32_16x16x32_bf16 (split-bf16, 3 terms) ----
  {
    const int w = t >> 5, lane = t & 31;
    const int half = lane >> 4, ln = lane & 15;
    const int rowb = (w & 1) * 16;       // waves 0,2 -> rows 0..15 ; 1,3 -> 16..31
    const int row  = rowb + ln;

    // A-frag (16-bit 16x32, contiguous order matching B): elem e -> K = kbase + half*16 + e
    BFrag ah0, ah1, al0, al1;
    #pragma unroll
    for (int e = 0; e < 16; ++e) {
      int kk = (half << 4) + e;
      ah0.u[e] = s_xhi[row * DIM + kk];
      ah1.u[e] = s_xhi[row * DIM + 32 + kk];
      al0.u[e] = s_xlo[row * DIM + kk];
      al1.u[e] = s_xlo[row * DIM + 32 + kk];
    }

    const int ct0 = (w & 2) ? 32 : 0;    // waves 0,1 -> tiles 0..31 ; 2,3 -> 32..63
    for (int ct = ct0; ct < ct0 + 32; ++ct) {
      const int col = ct * 16 + ln;
      // B-frag (16-bit 32x16, same order as A): elem e -> K = kbase + half*16 + e (contiguous)
      const size_t bb = (size_t)col * DIM + (half << 4);
      BFrag bh0, bh1, bl0, bl1;
      const v4u* ph0 = (const v4u*)(cbhi + bb);
      const v4u* ph1 = (const v4u*)(cbhi + bb + 32);
      const v4u* pl0 = (const v4u*)(cblo + bb);
      const v4u* pl1 = (const v4u*)(cblo + bb + 32);
      bh0.q[0] = ph0[0]; bh0.q[1] = ph0[1];
      bh1.q[0] = ph1[0]; bh1.q[1] = ph1[1];
      bl0.q[0] = pl0[0]; bl0.q[1] = pl0[1];
      bl1.q[0] = pl1[0]; bl1.q[1] = pl1[1];

      v8f acc = {};
      acc = __builtin_amdgcn_wmma_f32_16x16x32_bf16(false, ah0.v, false, bh0.v, (short)0, acc, false, false);
      acc = __builtin_amdgcn_wmma_f32_16x16x32_bf16(false, ah1.v, false, bh1.v, (short)0, acc, false, false);
      acc = __builtin_amdgcn_wmma_f32_16x16x32_bf16(false, ah0.v, false, bl0.v, (short)0, acc, false, false);
      acc = __builtin_amdgcn_wmma_f32_16x16x32_bf16(false, ah1.v, false, bl1.v, (short)0, acc, false, false);
      acc = __builtin_amdgcn_wmma_f32_16x16x32_bf16(false, al0.v, false, bh0.v, (short)0, acc, false, false);
      acc = __builtin_amdgcn_wmma_f32_16x16x32_bf16(false, al1.v, false, bh1.v, (short)0, acc, false, false);

      const float c2v = s_c2[col];
      #pragma unroll
      for (int v = 0; v < 8; ++v) {      // C/D layout: M = v + 8*half, N = ln
        int rl = rowb + v + (half << 3);
        s_dist[rl * DSTRIDE + col] = s_x2[rl] + c2v - 2.0f * acc[v];
      }
    }
  }
  __syncthreads();

  // ---- phase 3: per-row reductions over LDS distances (4 quarter-threads/row) ----
  const int r = t & (ROWS - 1);
  const int q = t >> 5;
  const int cbase = q * 256;
  const int dbase = r * DSTRIDE + cbase;

  // pass A: min / argmin (first-index tie-break, like jnp.argmin)
  {
    float best = 3.4e38f; int bi = cbase;
    for (int j = 0; j < 256; ++j) {
      float d = s_dist[dbase + j];
      if (d < best) { best = d; bi = cbase + j; }
    }
    s_ra[t] = best; s_ri[t] = bi;
  }
  __syncthreads();
  if (t < ROWS) {
    float best = s_ra[t]; int bi = s_ri[t];
    for (int qq = 1; qq < 4; ++qq) {
      float d = s_ra[t + 32 * qq]; int di = s_ri[t + 32 * qq];
      if (d < best || (d == best && di < bi)) { best = d; bi = di; }
    }
    s_tok[t]  = bi;
    s_amax[t] = -INV_TEMP * best;        // max affinity = -100 * min distance
  }
  __syncthreads();

  // pass B: e = exp(a - amax) written in place; accumulate S and sum(e*a)
  {
    const float am = s_amax[r];
    float ssum = 0.f, tsum = 0.f;
    for (int j = 0; j < 256; ++j) {
      float d = s_dist[dbase + j];
      float a = -INV_TEMP * d;
      float e = __expf(a - am);
      ssum += e;
      tsum = fmaf(e, a, tsum);
      s_dist[dbase + j] = e;
    }
    s_ra[t] = ssum; s_rb[t] = tsum;
  }
  __syncthreads();
  if (t < ROWS) {
    float S = 0.f, T = 0.f;
    for (int qq = 0; qq < 4; ++qq) { S += s_ra[t + 32 * qq]; T += s_rb[t + 32 * qq]; }
    s_invS[t] = 1.0f / S;
    // sum_k p*log p = sum_k p*a - lse = T/S - (amax + log S)
    s_entp[t] = T / S - (s_amax[t] + logf(S));
  }
  for (int i = t; i < KCB; i += THREADS) s_avgp[i] = 0.f;
  __syncthreads();

  // pass C: avg_probs accumulation (staggered to avoid LDS atomic contention)
  {
    const float is = s_invS[r];
    const int off = (r * 8) & 255;
    for (int j = 0; j < 256; ++j) {
      int jj = (j + off) & 255;
      atomicAdd(&s_avgp[cbase + jj], s_dist[dbase + jj] * is);
    }
  }
  __syncthreads();
  for (int i = t; i < KCB; i += THREADS)
    avgpart[(size_t)blk * KCB + i] = s_avgp[i];
  if (t == 0) {
    float es = 0.f;
    for (int i = 0; i < ROWS; ++i) es += s_entp[i];
    entpart[blk] = es;
  }

  // ---- phase 4: write out = x + (emb - x) and MSE partial ----
  {
    float m = 0.f;
    for (int i = t; i < ROWS * DIM; i += THREADS) {
      int rl = i >> 6, dd = i & 63;
      int tok = s_tok[rl];
      float c  = cbf32[(size_t)tok * DIM + dd];
      size_t gi = (size_t)(rowg0 + rl) * DIM + dd;
      float xv = xin[gi];
      out[gi] = xv + (c - xv);           // straight-through estimator, bit-exact form
      float df = xv - c;
      m = fmaf(df, df, m);
    }
    s_ra[t] = m;
  }
  __syncthreads();
  if (t == 0) {
    float ms = 0.f;
    for (int i = 0; i < THREADS; ++i) ms += s_ra[i];
    msepart[blk] = ms;
  }
}

// ============ kernel 3: deterministic final reduction -> scalar loss ============
__global__ void __launch_bounds__(1024)
vq_final(const float* __restrict__ avgpart, const float* __restrict__ entpart,
         const float* __restrict__ msepart, float* __restrict__ out_loss) {
  __shared__ float s_red[1024];
  const int k = threadIdx.x;
  float s = 0.f;
  for (int b = 0; b < NBLK; ++b) s += avgpart[(size_t)b * KCB + k];  // coalesced
  float ap = s * (1.0f / (float)NVEC);
  s_red[k] = ap * logf(ap + 1e-5f);
  __syncthreads();
  for (int stride = 512; stride > 0; stride >>= 1) {
    if (k < stride) s_red[k] += s_red[k + stride];
    __syncthreads();
  }
  if (k == 0) {
    float avg_entropy = -s_red[0];
    float es = 0.f, ms = 0.f;
    for (int b = 0; b < NBLK; ++b) { es += entpart[b]; ms += msepart[b]; }
    float sample_entropy = -es / (float)NVEC;
    float entropy_loss = sample_entropy - avg_entropy;
    float mse = ms / (float)(NVEC * DIM);
    out_loss[0] = 0.1f * entropy_loss + 1.25f * mse;  // COMMIT_W*mse + mse, ENTROPY_W*ent
  }
}

// ============ launcher ============
extern "C" void kernel_launch(void* const* d_in, const int* in_sizes, int n_in,
                              void* d_out, int out_size, void* d_ws, size_t ws_size,
                              hipStream_t stream) {
  (void)in_sizes; (void)n_in; (void)out_size; (void)ws_size;
  const float* x  = (const float*)d_in[0];   // [32,32,32,64] f32
  const float* cb = (const float*)d_in[1];   // [1024,64] f32
  float* outf = (float*)d_out;               // [N*64] out ++ [1] loss

  char* ws = (char*)d_ws;
  unsigned short* cbhi = (unsigned short*)(ws);                       // 128 KB
  unsigned short* cblo = (unsigned short*)(ws + 131072);              // 128 KB
  float* c2      = (float*)(ws + 262144);                             // 4 KB
  float* avgpart = (float*)(ws + 266240);                             // NBLK*KCB*4 = 4 MB
  float* entpart = (float*)(ws + 266240 + (size_t)NBLK * KCB * 4);    // 4 KB
  float* msepart = (float*)(ws + 266240 + (size_t)NBLK * KCB * 4 + NBLK * 4);

  vq_prep<<<(KCB + 255) / 256, 256, 0, stream>>>(cb, cbhi, cblo, c2);
  vq_main<<<NBLK, THREADS, 0, stream>>>(x, cb, cbhi, cblo, c2,
                                        outf, avgpart, entpart, msepart);
  vq_final<<<1, 1024, 0, stream>>>(avgpart, entpart, msepart,
                                   outf + (size_t)NVEC * DIM);
}
